// LlamaQAttention_47751446397245
// MI455X (gfx1250) — matrix operand order repaired and checked by reading, not compiled, not run on hardware
//
#include <hip/hip_runtime.h>

typedef __attribute__((ext_vector_type(16))) __bf16 v16bf;
typedef __attribute__((ext_vector_type(8)))  __bf16 v8bf;
typedef __attribute__((ext_vector_type(4)))  __bf16 v4bf;
typedef __attribute__((ext_vector_type(8)))  float  v8f;

#define S_LEN   2048
#define HID     4096
#define NHEADS  32
#define NKVH    8
#define DHEAD   128
#define QKV_OUT 6144   // (32 + 2*8) * 128

// ---------------------------------------------------------------------------
// WMMA helpers (CDNA5 wave32, V_WMMA_F32_16X16X32_BF16)
// ---------------------------------------------------------------------------
__device__ __forceinline__ v8f wmma_bf16f32(v16bf a, v16bf b, v8f c) {
  return __builtin_amdgcn_wmma_f32_16x16x32_bf16(false, a, false, b, (short)0, c,
                                                 false, false);
}

// A-matrix 16x32 bf16 fragment. Lane L (<16): M=L, K in {0..7, 16..23};
// lane L (>=16): M=L-16, K in {8..15, 24..31}.  base points at [M=0][K=0],
// ld = row stride in elements (must be mult of 8 for 16B alignment).
__device__ __forceinline__ v16bf frag_a(const __bf16* base, int ld, int lane) {
  const int row = lane & 15;
  const int kb  = (lane >> 4) << 3;
  const __bf16* p = base + row * ld + kb;
  v8bf lo = *(const v8bf*)p;
  v8bf hi = *(const v8bf*)(p + 16);
  v16bf r;
#pragma unroll
  for (int i = 0; i < 8; ++i) { r[i] = lo[i]; r[i + 8] = hi[i]; }
  return r;
}

// B-matrix 32x16 bf16 fragment. Lane L (<16): N=L, K in {0..7, 16..23};
// lane L (>=16): N=L-16, K in {8..15, 24..31}.  base points at [K=0][N=0], ld = row stride in elements.
__device__ __forceinline__ v16bf frag_b(const __bf16* base, int ld, int lane) {
  const int kb = (lane >> 4) << 3;
  const __bf16* p = base + (lane & 15);
  v16bf r;
#pragma unroll
  for (int i = 0; i < 8; ++i) { r[i] = p[(kb + i) * ld]; r[i + 8] = p[(16 + kb + i) * ld]; }
  return r;
}

// ---------------------------------------------------------------------------
// 1. f32 -> bf16 convert (vectorized x4)
// ---------------------------------------------------------------------------
__global__ void cvt_f32_bf16(const float* __restrict__ in,
                             __bf16* __restrict__ out, int n4) {
  int i = blockIdx.x * blockDim.x + threadIdx.x;
  if (i >= n4) return;
  float4 v = ((const float4*)in)[i];
  v4bf o;
  o[0] = (__bf16)v.x; o[1] = (__bf16)v.y; o[2] = (__bf16)v.z; o[3] = (__bf16)v.w;
  *(v4bf*)(out + (size_t)i * 4) = o;
}

// ---------------------------------------------------------------------------
// 2. Transpose + convert: in[R][C] f32 -> out[C][R] bf16 (32x32 LDS tiles)
// ---------------------------------------------------------------------------
__global__ void transpose_cvt(const float* __restrict__ in,
                              __bf16* __restrict__ out, int R, int C) {
  __shared__ float tile[32][33];
  const int c0 = blockIdx.x * 32, r0 = blockIdx.y * 32;
  const int tx = threadIdx.x, ty = threadIdx.y;  // 32 x 8
#pragma unroll
  for (int i = 0; i < 4; ++i)
    tile[ty + 8 * i][tx] = in[(size_t)(r0 + ty + 8 * i) * C + c0 + tx];
  __syncthreads();
#pragma unroll
  for (int i = 0; i < 4; ++i)
    out[(size_t)(c0 + ty + 8 * i) * R + r0 + tx] = (__bf16)tile[tx][ty + 8 * i];
}

// ---------------------------------------------------------------------------
// 3. Tiled bf16 WMMA GEMM: C[M][N] = A[M][K] * B[K][N]
//    128x128 block tile, 256 threads (8 waves), wave tile 32x64.
//    K-tile = 64 (16 WMMAs per barrier stage) + next-tile global prefetch.
// ---------------------------------------------------------------------------
template <int OUT_F32>
__global__ __launch_bounds__(256) void gemm_bt(const __bf16* __restrict__ A,
                                               const __bf16* __restrict__ B,
                                               void* __restrict__ Cp,
                                               int M, int N, int K) {
  __shared__ __bf16 As[128][72];   // 128 x 64 (+8 pad)
  __shared__ __bf16 Bs[64][136];   // 64 x 128 (+8 pad)
  const int tid = threadIdx.x, lane = tid & 31, wave = tid >> 5;
  const int wm = wave & 3, wn = wave >> 2;              // 4 x 2 wave grid
  const int mBase = blockIdx.y * 128, nBase = blockIdx.x * 128;
  const v8f zero = {0.f, 0.f, 0.f, 0.f, 0.f, 0.f, 0.f, 0.f};
  v8f acc[2][4];
#pragma unroll
  for (int i = 0; i < 2; ++i)
#pragma unroll
    for (int j = 0; j < 4; ++j) acc[i][j] = zero;

  const int arow = tid >> 1, acol = (tid & 1) * 32;     // A: 128 rows x 64
  const int brow = tid >> 2, bcol = (tid & 3) * 32;     // B: 64 rows x 128

  for (int k0 = 0; k0 < K; k0 += 64) {
    const __bf16* ap = A + (size_t)(mBase + arow) * K + k0 + acol;
#pragma unroll
    for (int u = 0; u < 4; ++u)
      *(v8bf*)&As[arow][acol + 8 * u] = *(const v8bf*)(ap + 8 * u);
    const __bf16* bp = B + (size_t)(k0 + brow) * N + nBase + bcol;
#pragma unroll
    for (int u = 0; u < 4; ++u)
      *(v8bf*)&Bs[brow][bcol + 8 * u] = *(const v8bf*)(bp + 8 * u);
    if (k0 + 64 < K) {   // prefetch next K-tile (global_prefetch_b8)
      __builtin_prefetch(ap + 64, 0, 1);
      __builtin_prefetch(bp + (size_t)64 * N, 0, 1);
    }
    __syncthreads();

#pragma unroll
    for (int kc = 0; kc < 2; ++kc) {
      v16bf af[2];
#pragma unroll
      for (int mt = 0; mt < 2; ++mt)
        af[mt] = frag_a(&As[wm * 32 + mt * 16][kc * 32], 72, lane);
#pragma unroll
      for (int nt = 0; nt < 4; ++nt) {
        v16bf bf = frag_b(&Bs[kc * 32][wn * 64 + nt * 16], 136, lane);
#pragma unroll
        for (int mt = 0; mt < 2; ++mt)
          acc[mt][nt] = wmma_bf16f32(af[mt], bf, acc[mt][nt]);
      }
    }
    __syncthreads();
  }

  const int hi = (lane >> 4) * 8, cn = lane & 15;
#pragma unroll
  for (int mt = 0; mt < 2; ++mt)
#pragma unroll
    for (int nt = 0; nt < 4; ++nt)
#pragma unroll
      for (int r = 0; r < 8; ++r) {
        const int row = mBase + wm * 32 + mt * 16 + r + hi;
        const int col = nBase + wn * 64 + nt * 16 + cn;
        const float v = acc[mt][nt][r];
        if (OUT_F32) ((float*)Cp)[(size_t)row * N + col] = v;
        else         ((__bf16*)Cp)[(size_t)row * N + col] = (__bf16)v;
      }
}

// ---------------------------------------------------------------------------
// 4. RoPE + GQA split: xqkv[S][6144] bf16 -> qT[32][S][128], kT[8][S][128],
//    vT[8][S][128] (head-major, bf16).
// ---------------------------------------------------------------------------
__global__ void rope_split(const __bf16* __restrict__ xqkv,
                           __bf16* __restrict__ qT, __bf16* __restrict__ kT,
                           __bf16* __restrict__ vT) {
  int idx = blockIdx.x * blockDim.x + threadIdx.x;
  if (idx >= S_LEN * 48 * 64) return;
  const int i = idx & 63;
  const int t = idx >> 6;
  const int head = t % 48;
  const int s = t / 48;
  const __bf16* src = xqkv + (size_t)s * QKV_OUT + head * DHEAD;
  if (head < NHEADS + NKVH) {
    const float xr = (float)src[i], xi = (float)src[i + 64];
    const float freq = __powf(10000.f, -(float)i * (1.f / 64.f));
    float sn, cs;
    __sincosf((float)s * freq, &sn, &cs);
    __bf16* dst = (head < NHEADS)
                      ? (qT + ((size_t)head * S_LEN + s) * DHEAD)
                      : (kT + ((size_t)(head - NHEADS) * S_LEN + s) * DHEAD);
    dst[i]      = (__bf16)(xr * cs - xi * sn);
    dst[i + 64] = (__bf16)(xr * sn + xi * cs);
  } else {
    __bf16* dst = vT + ((size_t)(head - NHEADS - NKVH) * S_LEN + s) * DHEAD;
    dst[i] = src[i];
    dst[i + 64] = src[i + 64];
  }
}

// ---------------------------------------------------------------------------
// 5. Flash attention (causal, GQA 4:1). Block = 1 head x 128 query rows,
//    8 waves x 16 rows each. 32-key tiles, online softmax, WMMA bf16.
// ---------------------------------------------------------------------------
__global__ __launch_bounds__(256) void flash_attn(const __bf16* __restrict__ qT,
                                                  const __bf16* __restrict__ kT,
                                                  const __bf16* __restrict__ vT,
                                                  __bf16* __restrict__ attn) {
  __shared__ __bf16 Kt[128][40];     // K^T: [d][key] for 32 keys
  __shared__ __bf16 Vs[32][136];     // V:   [key][d]
  __shared__ __bf16 Pl[8][16][40];   // per-wave P re-layout buffer
  const int tid = threadIdx.x, lane = tid & 31, wave = tid >> 5;
  const int h = blockIdx.y, kvh = h >> 2;
  const int qBase = blockIdx.x * 128;
  const int qRow = qBase + wave * 16;
  const int hi = lane >> 4, cn = lane & 15;
  const v8f zero = {0.f, 0.f, 0.f, 0.f, 0.f, 0.f, 0.f, 0.f};

  // Resident Q fragments (16 rows x 128 d = 4 chunks of K=32)
  const __bf16* qp = qT + ((size_t)h * S_LEN + qRow) * DHEAD;
  v16bf qf[4];
#pragma unroll
  for (int c = 0; c < 4; ++c) qf[c] = frag_a(qp + c * 32, DHEAD, lane);

  v8f o[8];
#pragma unroll
  for (int t = 0; t < 8; ++t) o[t] = zero;
  float m[8], l[8];
#pragma unroll
  for (int r = 0; r < 8; ++r) { m[r] = -1e30f; l[r] = 0.f; }

  const __bf16* kbasep = kT + (size_t)kvh * S_LEN * DHEAD;
  const __bf16* vbasep = vT + (size_t)kvh * S_LEN * DHEAD;
  const int ckey = tid >> 3, cseg = (tid & 7) * 16;
  const int kEnd = qBase + 128;
  const float scale = 0.08838834764831845f;  // 1/sqrt(128)

  for (int kb = 0; kb < kEnd; kb += 32) {
    const __bf16* ks = kbasep + (size_t)(kb + ckey) * DHEAD + cseg;
    const __bf16* vs = vbasep + (size_t)(kb + ckey) * DHEAD + cseg;
    if (kb + 32 < kEnd) {   // prefetch next K/V tile
      __builtin_prefetch(ks + 32 * DHEAD, 0, 1);
      __builtin_prefetch(vs + 32 * DHEAD, 0, 1);
    }
    __syncthreads();  // previous tile fully consumed
    // Stage K transposed ([d][key]) and V ([key][d]) cooperatively.
#pragma unroll
    for (int j = 0; j < 16; ++j) Kt[cseg + j][ckey] = ks[j];
    *(v8bf*)&Vs[ckey][cseg]     = *(const v8bf*)vs;
    *(v8bf*)&Vs[ckey][cseg + 8] = *(const v8bf*)(vs + 8);
    __syncthreads();

    // Scores: S[16q][32k] = Q(16x128) @ K^T(128x32), two 16-col accumulators.
    v8f s0 = zero, s1 = zero;
#pragma unroll
    for (int c = 0; c < 4; ++c) {
      s0 = wmma_bf16f32(qf[c], frag_b(&Kt[c * 32][0], 40, lane), s0);
      s1 = wmma_bf16f32(qf[c], frag_b(&Kt[c * 32][16], 40, lane), s1);
    }

    // Online softmax (rows live across 16-lane halves; xor-shuffle reduce).
#pragma unroll
    for (int r = 0; r < 8; ++r) {
      const int qi = qRow + r + hi * 8;
      float x0 = s0[r] * scale + ((kb + cn)      > qi ? -1e9f : 0.f);
      float x1 = s1[r] * scale + ((kb + 16 + cn) > qi ? -1e9f : 0.f);
      float mx = fmaxf(x0, x1);
#pragma unroll
      for (int msk = 1; msk <= 8; msk <<= 1)
        mx = fmaxf(mx, __shfl_xor(mx, msk, 32));
      const float mnew = fmaxf(m[r], mx);
      const float alpha = __expf(m[r] - mnew);
      const float p0 = __expf(x0 - mnew), p1 = __expf(x1 - mnew);
      float rs = p0 + p1;
#pragma unroll
      for (int msk = 1; msk <= 8; msk <<= 1) rs += __shfl_xor(rs, msk, 32);
      l[r] = l[r] * alpha + rs;
      m[r] = mnew;
#pragma unroll
      for (int t = 0; t < 8; ++t) o[t][r] *= alpha;
      Pl[wave][r + hi * 8][cn]      = (__bf16)p0;   // re-layout P via LDS
      Pl[wave][r + hi * 8][16 + cn] = (__bf16)p1;
    }

    // O += P(16x32) @ V(32x128)  (per-wave LDS region; same-wave DS in-order)
    v16bf pa = frag_a(&Pl[wave][0][0], 40, lane);
#pragma unroll
    for (int t = 0; t < 8; ++t)
      o[t] = wmma_bf16f32(pa, frag_b(&Vs[0][t * 16], 136, lane), o[t]);
  }

  // Epilogue: normalize and scatter into attn[S][NH*D] (bf16).
#pragma unroll
  for (int r = 0; r < 8; ++r) {
    const float inv = 1.f / l[r];
    const int q = qRow + r + hi * 8;
#pragma unroll
    for (int t = 0; t < 8; ++t)
      attn[(size_t)q * HID + h * DHEAD + t * 16 + cn] = (__bf16)(o[t][r] * inv);
  }
}

// ---------------------------------------------------------------------------
// Launch
// ---------------------------------------------------------------------------
extern "C" void kernel_launch(void* const* d_in, const int* in_sizes, int n_in,
                              void* d_out, int out_size, void* d_ws,
                              size_t ws_size, hipStream_t stream) {
  const float* hs   = (const float*)d_in[0];  // [1,2048,4096]
  // d_in[1] = attention_mask: pure causal, applied analytically in-kernel.
  const float* wqkv = (const float*)d_in[2];  // [6144,4096]
  const float* wo   = (const float*)d_in[3];  // [4096,4096]

  char* ws = (char*)d_ws;
  size_t off = 0;
  __bf16* hsb   = (__bf16*)(ws + off); off += (size_t)S_LEN * HID * 2;
  __bf16* wqkvT = (__bf16*)(ws + off); off += (size_t)HID * QKV_OUT * 2;
  __bf16* woT   = (__bf16*)(ws + off); off += (size_t)HID * HID * 2;
  __bf16* xqkvb = (__bf16*)(ws + off); off += (size_t)S_LEN * QKV_OUT * 2;
  __bf16* qTb   = (__bf16*)(ws + off); off += (size_t)NHEADS * S_LEN * DHEAD * 2;
  __bf16* kTb   = (__bf16*)(ws + off); off += (size_t)NKVH * S_LEN * DHEAD * 2;
  __bf16* vTb   = (__bf16*)(ws + off); off += (size_t)NKVH * S_LEN * DHEAD * 2;
  __bf16* attnb = (__bf16*)(ws + off); off += (size_t)S_LEN * HID * 2;

  cvt_f32_bf16<<<(S_LEN * HID / 4 + 255) / 256, 256, 0, stream>>>(
      hs, hsb, S_LEN * HID / 4);
  transpose_cvt<<<dim3(HID / 32, QKV_OUT / 32), dim3(32, 8), 0, stream>>>(
      wqkv, wqkvT, QKV_OUT, HID);
  transpose_cvt<<<dim3(HID / 32, HID / 32), dim3(32, 8), 0, stream>>>(
      wo, woT, HID, HID);
  gemm_bt<0><<<dim3(QKV_OUT / 128, S_LEN / 128), 256, 0, stream>>>(
      hsb, wqkvT, (void*)xqkvb, S_LEN, QKV_OUT, HID);
  rope_split<<<(S_LEN * 48 * 64 + 255) / 256, 256, 0, stream>>>(
      xqkvb, qTb, kTb, vTb);
  flash_attn<<<dim3(S_LEN / 128, NHEADS), 256, 0, stream>>>(
      qTb, kTb, vTb, attnb);
  gemm_bt<1><<<dim3(HID / 128, S_LEN / 128), 256, 0, stream>>>(
      attnb, woT, d_out, S_LEN, HID, HID);
}
